// ConvSE3_89928025244022
// MI455X (gfx1250) — matrix operand order repaired and checked by reading, not compiled, not run on hardware
//
#include <hip/hip_runtime.h>

// ConvSE3 (SE3-Transformer conv layer) fused kernel for MI455X / gfx1250.
// Radial MLP 1->128->128->f*256 per edge (~17 GFLOP) on v_wmma_f32_16x16x32_f16.
// Round 2: weights pre-converted to f16 and pre-swizzled into B-fragment order
// in d_ws (one small pre-pass kernel); activations stored A-fragment-swizzled
// in LDS. Fragments load as contiguous 32B per lane (b128 pairs).

typedef __attribute__((ext_vector_type(16))) _Float16 v16h;
typedef __attribute__((ext_vector_type(8)))  float    v8f;

#define Bv   4
#define Nv   256
#define Kv   32
#define MIDv 128
#define EPSv 1e-5f

// ---- params buffer layout (jax pytree leaf order: sorted dict keys) ----
// per pair: b1[128] b2[128] b3[f*256] be1[128] be2[128] g1[128] g2[128]
//           w1[128] w2[128*128] w3[128*f*256]
#define POFF_00 0
#define POFF_01 50304
#define POFF_10 100608
#define POFF_11 150912
#define POFF_SI0 267264
#define POFF_SI1 267520

// ---- f16 swizzled weight workspace layout (element = _Float16) ----
// w2[pk] at pk*16384 ; w3[pk] at {65536, 98304, 131072, 163840}
#define WS_W2(pk)  ((pk) * 16384)
__device__ __host__ inline int ws_w3(int pk) {
  const int off[4] = {65536, 98304, 131072, 163840};
  return off[pk];
}

// B-matrix mapping (32x16 f16): col = nt*16 + (lane&15); K = 32*ks + 8*(lane>>4) + (h<8 ? h : h+8).
// Swizzled index so each lane's 16 halves are contiguous (32B).
__device__ __host__ inline int swz_idx(int K, int col) {
  const int ks   = K >> 5;
  const int g    = (K >> 3) & 1;
  const int h    = (K & 7) | ((K >> 1) & 8);
  const int lane = (col & 15) + (g << 4);
  const int nt   = col >> 4;
  return ((((nt << 2) + ks) * 32 + lane) << 4) + h;
}

// ---------------- pre-pass: convert f32 weights -> swizzled f16 in ws --------
__global__ __launch_bounds__(256)
void convert_weights(const float* __restrict__ params, _Float16* __restrict__ ws) {
  const int pk = blockIdx.y;
  const int f  = (pk == 3) ? 3 : 1;
  const int pairOff[4] = {POFF_00, POFF_01, POFF_10, POFF_11};
  const int ncols3 = 256 * f;
  const int n_w2 = MIDv * MIDv;
  const int n_w3 = MIDv * ncols3;
  const float* P  = params + pairOff[pk];
  const float* w2 = P + 896 + 256 * f;
  const float* w3 = w2 + n_w2;
  _Float16* d2 = ws + WS_W2(pk);
  _Float16* d3 = ws + ws_w3(pk);
  for (int j = blockIdx.x * blockDim.x + threadIdx.x; j < n_w2 + n_w3;
       j += gridDim.x * blockDim.x) {
    if (j < n_w2) {
      const int K = j >> 7, col = j & 127;
      d2[swz_idx(K, col)] = (_Float16)w2[j];
    } else {
      const int jj = j - n_w2;
      const int K = jj / ncols3, col = jj - K * ncols3;
      d3[swz_idx(K, col)] = (_Float16)w3[jj];
    }
  }
}

// ---------------- fragment loads (contiguous 32B per lane) ------------------
__device__ inline v16h load_A_swz(const _Float16* hA, int mtile, int ks, int lane) {
  return *(const v16h*)(hA + ((((mtile << 2) + ks) * 32 + lane) << 4));
}
__device__ inline v16h load_B_swz(const _Float16* __restrict__ w, int nt, int ks, int lane) {
  return *(const v16h*)(w + ((((nt << 2) + ks) * 32 + lane) << 4));
}

// LayerNorm(128)+ReLU: f32 LDS row-major -> f16 LDS in A-fragment swizzled order.
// A-matrix (16x32 f16, ISA 7.12.2): row = lane&15 (+16*mtile); within a 32-chunk,
// K = kb+(h&1)+8g with kb=2v (v<4) / 16+2(v-4) (v>=4) -> K_local = h (h<8), h+8 (h>=8).
__device__ inline void ln_relu_swz(const float* pre, _Float16* dstA,
                                   const float* __restrict__ gamma,
                                   const float* __restrict__ beta,
                                   float* s_mu, float* s_rstd, int t) {
  if (t < 32) {
    float s = 0.f, s2 = 0.f;
    const float* row = pre + t * MIDv;
#pragma unroll 8
    for (int c = 0; c < MIDv; ++c) { const float x = row[c]; s += x; s2 += x * x; }
    const float mu  = s * (1.f / MIDv);
    const float var = s2 * (1.f / MIDv) - mu * mu;
    s_mu[t]   = mu;
    s_rstd[t] = rsqrtf(var + EPSv);
  }
  __syncthreads();
  for (int idx = t; idx < 32 * MIDv; idx += 256) {
    const int k = idx >> 7, c = idx & (MIDv - 1);
    const float h = (pre[idx] - s_mu[k]) * s_rstd[k] * gamma[c] + beta[c];
    const int ks = c >> 5;
    const int kk = c & 31;
    const int g  = (kk >> 3) & 1;
    const int kl = kk - (g << 3);            // in {0..7} u {16..23}
    const int hh = (kl < 8) ? kl : (kl - 8);
    const int lane = (k & 15) + (g << 4);
    const int mt = k >> 4;
    dstA[((((mt << 2) + ks) * 32 + lane) << 4) + hh] = (_Float16)fmaxf(h, 0.f);
  }
  __syncthreads();
}

__global__ __launch_bounds__(256)
void conv_se3_fused(const float* __restrict__ x0,  const float* __restrict__ x1,
                    const float* __restrict__ rel, const float* __restrict__ bas00,
                    const float* __restrict__ bas01,const float* __restrict__ bas10,
                    const float* __restrict__ bas11,const float* __restrict__ params,
                    const int* __restrict__ nbr,   const int* __restrict__ msk,
                    const _Float16* __restrict__ wsw, float* __restrict__ out) {
  __shared__ float    s_pre[32 * MIDv];                       // 16 KB
  __shared__ __attribute__((aligned(32))) _Float16 s_hA1[32 * MIDv]; // 8 KB, swizzled
  __shared__ __attribute__((aligned(32))) _Float16 s_hA2[32 * MIDv]; // 8 KB, swizzled
  __shared__ float    s_x0g[32 * 16];
  __shared__ float    s_x1g[32 * 48];
  __shared__ float    s_b00[32];
  __shared__ float    s_b01[96];
  __shared__ float    s_b10[96];
  __shared__ float    s_b11[32 * 27];
  __shared__ float    s_acc0[32 * 16];
  __shared__ float    s_acc1[32 * 48];
  __shared__ float    s_feat[32], s_mask[32], s_mu[32], s_rstd[32];
  __shared__ int      s_nbr[32];

  const int t    = threadIdx.x;
  const int lane = t & 31;
  const int wave = t >> 5;
  const int node = blockIdx.x;               // flat b*N + n
  const int b    = node / Nv;

  // ---------------- gather neighbors / basis / accumulator init -------------
  if (t < 32) {
    const int e = node * Kv + t;
    s_nbr[t]  = nbr[e];
    s_feat[t] = rel[e];
    s_mask[t] = (float)msk[e];
  }
  __syncthreads();
  for (int idx = t; idx < 32 * 16; idx += 256) {
    const int k = idx >> 4, d = idx & 15;
    s_x0g[idx] = x0[(b * Nv + s_nbr[k]) * 16 + d];
  }
  for (int idx = t; idx < 32 * 48; idx += 256) {
    const int k = idx / 48, r = idx - k * 48;
    s_x1g[idx] = x1[(b * Nv + s_nbr[k]) * 48 + r];
  }
  for (int idx = t; idx < 32;      idx += 256) s_b00[idx] = bas00[node * Kv      + idx];
  for (int idx = t; idx < 96;      idx += 256) s_b01[idx] = bas01[node * Kv * 3  + idx];
  for (int idx = t; idx < 96;      idx += 256) s_b10[idx] = bas10[node * Kv * 3  + idx];
  for (int idx = t; idx < 32 * 27; idx += 256) s_b11[idx] = bas11[node * Kv * 27 + idx];
  for (int idx = t; idx < 32 * 16; idx += 256) s_acc0[idx] = 0.f;
  for (int idx = t; idx < 32 * 48; idx += 256) s_acc1[idx] = 0.f;
  __syncthreads();

  const int pairOff[4] = {POFF_00, POFF_01, POFF_10, POFF_11};

  for (int pk = 0; pk < 4; ++pk) {
    const int f = (pk == 3) ? 3 : 1;
    const float* P   = params + pairOff[pk];
    const float* b1  = P;
    const float* b2  = P + 128;
    const float* b3  = P + 256;
    const float* be1 = P + 256 + 256 * f;
    const float* be2 = be1 + 128;
    const float* g1  = be2 + 128;
    const float* g2  = g1 + 128;
    const float* w1  = g2 + 128;
    const _Float16* w2s = wsw + WS_W2(pk);
    const _Float16* w3s = wsw + ws_w3(pk);
    __builtin_prefetch(w2s, 0, 1);               // global_prefetch_b8
    __builtin_prefetch(w3s, 0, 1);

    // ---- layer 1: rank-1 (scalar dist) -> LN -> ReLU -> f16 (swizzled) ----
    for (int idx = t; idx < 32 * MIDv; idx += 256) {
      const int k = idx >> 7, c = idx & 127;
      s_pre[idx] = s_feat[k] * w1[c] + b1[c];
    }
    __syncthreads();
    ln_relu_swz(s_pre, s_hA1, g1, be1, s_mu, s_rstd, t);

    // ---- layer 2: [32,128] x [128,128] via WMMA (wave w owns N-tile w) ----
    {
      v16h a0[4], a1[4];
#pragma unroll
      for (int ks = 0; ks < 4; ++ks) {
        a0[ks] = load_A_swz(s_hA1, 0, ks, lane);
        a1[ks] = load_A_swz(s_hA1, 1, ks, lane);
      }
      v8f c0 = {}; v8f c1 = {};
#pragma unroll
      for (int ks = 0; ks < 4; ++ks) {
        const v16h bf = load_B_swz(w2s, wave, ks, lane);
        c0 = __builtin_amdgcn_wmma_f32_16x16x32_f16(false, a0[ks], false, bf, (short)0, c0, false, false);
        c1 = __builtin_amdgcn_wmma_f32_16x16x32_f16(false, a1[ks], false, bf, (short)0, c1, false, false);
      }
      const int col   = (wave << 4) + (lane & 15);
      const float bia = b2[col];
      const int rbase = (lane >> 4) << 3;          // D rows: r + 8*(lane>>4)
#pragma unroll
      for (int r = 0; r < 8; ++r) {
        s_pre[(rbase + r)      * MIDv + col] = c0[r] + bia;
        s_pre[(rbase + r + 16) * MIDv + col] = c1[r] + bia;
      }
    }
    __syncthreads();
    ln_relu_swz(s_pre, s_hA2, g2, be2, s_mu, s_rstd, t);

    // ---- layer 3: [32,128] x [128, f*256] WMMA + fused basis contraction ----
    v16h a0[4], a1[4];
#pragma unroll
    for (int ks = 0; ks < 4; ++ks) {
      a0[ks] = load_A_swz(s_hA2, 0, ks, lane);
      a1[ks] = load_A_swz(s_hA2, 1, ks, lane);
    }
    for (int it = 0; it < 2 * f; ++it) {           // 16f N-tiles over 8 waves
      const int nt = wave + (it << 3);
      v8f c0 = {}; v8f c1 = {};
#pragma unroll
      for (int ks = 0; ks < 4; ++ks) {
        const v16h bf = load_B_swz(w3s, nt, ks, lane);
        c0 = __builtin_amdgcn_wmma_f32_16x16x32_f16(false, a0[ks], false, bf, (short)0, c0, false, false);
        c1 = __builtin_amdgcn_wmma_f32_16x16x32_f16(false, a1[ks], false, bf, (short)0, c1, false, false);
      }
      // lane owns one output column = fixed (o, i, fi); 8 edges per M-tile.
      const int col  = (nt << 4) + (lane & 15);
      const float bia = b3[col];
      const int o   = col / (16 * f);
      const int i   = (col / f) & 15;
      const int fi  = col - (col / f) * f;
      const int rbase = (lane >> 4) << 3;
#pragma unroll
      for (int r = 0; r < 8; ++r) {
#pragma unroll
        for (int mt = 0; mt < 2; ++mt) {
          const int   e = rbase + r + (mt << 4);
          const float v = ((mt == 0) ? c0[r] : c1[r]) + bia;
          if (pk == 0) {            // '0,0' -> acc0
            atomicAdd(&s_acc0[e * 16 + o], v * s_b00[e] * s_x0g[e * 16 + i]);
          } else if (pk == 1) {     // '0,1' -> acc1
            const float xv = s_x0g[e * 16 + i];
#pragma unroll
            for (int p = 0; p < 3; ++p)
              atomicAdd(&s_acc1[(e * 16 + o) * 3 + p], v * s_b01[e * 3 + p] * xv);
          } else if (pk == 2) {     // '1,0' -> acc0
            float s = 0.f;
#pragma unroll
            for (int q = 0; q < 3; ++q)
              s += s_b10[e * 3 + q] * s_x1g[(e * 16 + i) * 3 + q];
            atomicAdd(&s_acc0[e * 16 + o], v * s);
          } else {                  // '1,1' -> acc1
#pragma unroll
            for (int p = 0; p < 3; ++p) {
              float s = 0.f;
#pragma unroll
              for (int q = 0; q < 3; ++q)
                s += s_b11[e * 27 + p * 9 + q * 3 + fi] * s_x1g[(e * 16 + i) * 3 + q];
              atomicAdd(&s_acc1[(e * 16 + o) * 3 + p], v * s);
            }
          }
        }
      }
    }
    __syncthreads();
  }

  // ---------------- masked mean over K + self-interaction -------------------
  float wsum = 0.f;
  for (int e = 0; e < 32; ++e) wsum += s_mask[e];
  const float denom = fmaxf(wsum, 1e-5f);
  const float* wsi0 = params + POFF_SI0;
  const float* wsi1 = params + POFF_SI1;
  if (t < 16) {
    float s = 0.f;
    for (int e = 0; e < 32; ++e) s += s_acc0[e * 16 + t] * s_mask[e];
    s /= denom;
    const float* xs0 = x0 + node * 16;
    float si = 0.f;
#pragma unroll
    for (int d = 0; d < 16; ++d) si += xs0[d] * wsi0[d * 16 + t];
    out[node * 16 + t] = s + si;
  }
  if (t < 48) {
    const int o = t / 3, p = t - o * 3;
    float s = 0.f;
    for (int e = 0; e < 32; ++e) s += s_acc1[(e * 16 + o) * 3 + p] * s_mask[e];
    s /= denom;
    const float* xs1 = x1 + node * 48;
    float si = 0.f;
#pragma unroll
    for (int d = 0; d < 16; ++d) si += xs1[d * 3 + p] * wsi1[d * 16 + o];
    out[Bv * Nv * 16 + node * 48 + o * 3 + p] = s + si;
  }
}

extern "C" void kernel_launch(void* const* d_in, const int* in_sizes, int n_in,
                              void* d_out, int out_size, void* d_ws, size_t ws_size,
                              hipStream_t stream) {
  const float* x0    = (const float*)d_in[0];
  const float* x1    = (const float*)d_in[1];
  const float* rel   = (const float*)d_in[2];
  const float* b00   = (const float*)d_in[3];
  const float* b01   = (const float*)d_in[4];
  const float* b10   = (const float*)d_in[5];
  const float* b11   = (const float*)d_in[6];
  const float* prm   = (const float*)d_in[7];
  const int*   nbr   = (const int*)d_in[8];
  const int*   msk   = (const int*)d_in[9];
  _Float16*    wsw   = (_Float16*)d_ws;     // 262144 halves = 512 KB
  float*       out   = (float*)d_out;
  (void)in_sizes; (void)n_in; (void)out_size; (void)ws_size;

  hipLaunchKernelGGL(convert_weights, dim3(64, 4), dim3(256), 0, stream, prm, wsw);
  hipLaunchKernelGGL(conv_se3_fused, dim3(Bv * Nv), dim3(256), 0, stream,
                     x0, x1, rel, b00, b01, b10, b11, prm, nbr, msk,
                     (const _Float16*)wsw, out);
}
